// DaRnnEncoder_76845554860770
// MI455X (gfx1250) — matrix operand order repaired and checked by reading, not compiled, not run on hardware
//
#include <hip/hip_runtime.h>
#include <hip/hip_bf16.h>
#include <math.h>

typedef __attribute__((ext_vector_type(16))) __bf16 v16bf;
typedef __attribute__((ext_vector_type(8)))  __bf16 v8bf;
typedef __attribute__((ext_vector_type(8)))  float  v8f;
typedef __attribute__((ext_vector_type(4)))  unsigned int v4u;
typedef __attribute__((ext_vector_type(8)))  int v8i;
typedef __attribute__((ext_vector_type(4)))  int v4i;

#define B_   256
#define T1_  255
#define N_   128
#define H_   256
#define K_   384      /* N_ + H_ : combined GEMM K */
#define NWG  16
#define NTHR 256
#define AST  392      /* padded row stride in bf16 elems (384 + 8) */
#define GCOLS 64      /* gate columns per WG: 4 gates x 16 h-cols  */

/* ---- workspace layout (bytes) ---- */
#define WS_WT    0                          /* Wt bf16 [1024][384]        */
#define WS_SX    (WS_WT + 1024*K_*2)        /* score_x f32 [256][128]     */
#define WS_HBF   (WS_SX + B_*N_*4)          /* h bf16 [2][256][256]       */
#define WS_SPART (WS_HBF + 2*B_*H_*2)       /* s partials f32 [2][16][256]*/
#define WS_CNT   (WS_SPART + 2*NWG*B_*4)    /* barrier counter            */
#define WS_END   (WS_CNT + 64)

__global__ void darnn_prep(const float* __restrict__ input,
                           const float* __restrict__ attn_W,
                           const float* __restrict__ attn_b,
                           const float* __restrict__ W_ih,
                           const float* __restrict__ W_hh,
                           unsigned char* __restrict__ ws) {
  __bf16* Wt = (__bf16*)(ws + WS_WT);
  float*  sx = (float*)(ws + WS_SX);
  const int stride = gridDim.x * blockDim.x;
  const int idx = blockIdx.x * blockDim.x + threadIdx.x;

  /* weights -> bf16, layout [gate_col][k] with k = [w_in K | h K] */
  for (int i = idx; i < 1024 * K_; i += stride) {
    int col = i / K_, k = i % K_;
    float v = (k < N_) ? W_ih[col * N_ + k] : W_hh[col * H_ + (k - N_)];
    Wt[i] = (__bf16)v;
  }
  /* score_x[b][n] = sum_t input[b,t,n] * Wx[t] + attn_b */
  for (int i = idx; i < B_ * N_; i += stride) {
    int b = i >> 7, n = i & 127;
    float acc = attn_b[0];
    const float* p = input + (size_t)b * T1_ * N_ + n;
    for (int t = 0; t < T1_; ++t) acc += p[(size_t)t * N_] * attn_W[2 * H_ + t];
    sx[i] = acc;
  }
  /* zero h-state, s-partials, barrier counter (fresh every launch) */
  unsigned* z = (unsigned*)(ws + WS_HBF);
  const int zwords = (WS_END - WS_HBF) / 4;
  for (int i = idx; i < zwords; i += stride) z[i] = 0u;
}

__global__ __launch_bounds__(NTHR, 1)
void darnn_main(const float* __restrict__ input,
                const float* __restrict__ attn_W,
                const float* __restrict__ b_ih,
                const float* __restrict__ b_hh,
                unsigned char* __restrict__ ws,
                float* __restrict__ out) {
  __shared__ __align__(16) __bf16 A_lds[B_ * AST];     /* [256][392] w_in|h */
  __shared__ __align__(16) __bf16 W_lds[GCOLS * AST];  /* [64][392]         */
  __shared__ float s_lds[B_];
  __shared__ float red_lds[2 * N_];

  const __bf16* Wt   = (const __bf16*)(ws + WS_WT);
  const float*  sx   = (const float*)(ws + WS_SX);
  __bf16*       hbf  = (__bf16*)(ws + WS_HBF);
  float*        spart= (float*)(ws + WS_SPART);
  unsigned*     cnt  = (unsigned*)(ws + WS_CNT);

  const int tid  = threadIdx.x;
  const int wg   = blockIdx.x;
  const int wave = tid >> 5, lane = tid & 31;
  const int lhalf = lane >> 4, l16 = lane & 15;

  /* load this WG's weight slice: local col lc -> gate g = lc/16, j = lc%16 */
  for (int i = tid; i < GCOLS * (K_ / 8); i += NTHR) {
    int lc = i / (K_ / 8);
    int kc = (i % (K_ / 8)) * 8;
    int g = lc >> 4, jj = lc & 15;
    int gcol = g * H_ + wg * 16 + jj;
    *(v8bf*)(&W_lds[lc * AST + kc]) = *(const v8bf*)(Wt + (size_t)gcol * K_ + kc);
  }

  /* per-lane constants for the owned h-column jglob = wg*16 + l16 */
  const int jglob = wg * 16 + l16;
  float bsum[4];
#pragma unroll
  for (int g = 0; g < 4; ++g) bsum[g] = b_ih[g * H_ + jglob] + b_hh[g * H_ + jglob];
  const float Whj = attn_W[jglob];
  const float Wcj = attn_W[H_ + jglob];

  float cReg[2][8];
#pragma unroll
  for (int m = 0; m < 2; ++m)
#pragma unroll
    for (int r = 0; r < 8; ++r) cReg[m][r] = 0.f;

  float* out_w = out;                           /* (B,T1,N) */
  float* out_e = out + (size_t)B_ * T1_ * N_;   /* (B,T1,H) */

  for (int t = 0; t < T1_; ++t) {
    const int par = t & 1, nxt = par ^ 1;

    /* prefetch next step's x_t row for batch b = tid */
    if (t + 1 < T1_)
      __builtin_prefetch(input + ((size_t)tid * T1_ + (t + 1)) * N_, 0, 1);

    /* ---- async: TDM pulls shared bf16 h -> A columns [128,384) ---- */
#if __has_builtin(__builtin_amdgcn_tensor_load_to_lds)
    if (wave == 0) {
      /* D# group0: count=1 | lds_addr | global_addr | type=2            */
      unsigned long long gaddr =
          (unsigned long long)(uintptr_t)(hbf + (size_t)par * B_ * H_);
      unsigned lds_addr = (unsigned)(uintptr_t)(&A_lds[N_]); /* low32 = LDS off */
      v4u g0;
      g0[0] = 1u;                                  /* count=1, user mode   */
      g0[1] = lds_addr;
      g0[2] = (unsigned)(gaddr & 0xffffffffu);
      g0[3] = (unsigned)((gaddr >> 32) & 0x01ffffffu) | 0x80000000u; /* type=2 */
      /* D# group1: data_size=2B, pad_enable, interval=128DW, amount=68DW */
      v8i g1;
      g1[0] = (int)((1u << 16) | (1u << 20) | (6u << 22) | (67u << 25));
      g1[1] = (int)(256u << 16);   /* tensor_dim0 = 256 (low16 @ bit48)   */
      g1[2] = (int)(256u << 16);   /* dim0 hi=0; tensor_dim1=256 lo16     */
      g1[3] = (int)(256u << 16);   /* dim1 hi=0; tile_dim0 = 256          */
      g1[4] = (int)256;            /* tile_dim1 = 256; tile_dim2 = 0      */
      g1[5] = (int)256;            /* tensor_dim0_stride = 256 (low32)    */
      g1[6] = 0;
      g1[7] = 0;
      v4i zz; zz[0] = 0; zz[1] = 0; zz[2] = 0; zz[3] = 0;
      v8i z8; z8[0] = 0; z8[1] = 0; z8[2] = 0; z8[3] = 0;
      z8[4] = 0; z8[5] = 0; z8[6] = 0; z8[7] = 0;
      /* amdgpu-toolchain (clang-23) 6-arg form */
      __builtin_amdgcn_tensor_load_to_lds(g0, g1, zz, zz, z8, 0);
    }
#else
    {
      const __bf16* hsrc = hbf + (size_t)par * B_ * H_;
      for (int i = tid; i < B_ * (H_ / 8); i += NTHR) {
        int row = i >> 5;
        int col = (i & 31) * 8;
        *(v8bf*)(&A_lds[row * AST + N_ + col]) = *(const v8bf*)(hsrc + (size_t)row * H_ + col);
      }
    }
#endif

    /* (a) gather s[b] from per-WG partials (deterministic fixed order) */
    {
      float s = 0.f;
      const float* sp = spart + (size_t)par * NWG * B_;
#pragma unroll
      for (int w = 0; w < NWG; ++w) s += sp[w * B_ + tid];
      s_lds[tid] = s;
    }
    __syncthreads();

    /* (b) softmax over batch per column n; w_in -> A[:,0:128) and d_out */
    {
      const int n = tid & 127, half = tid >> 7;
      const int b0 = half * 128;
      float m = -1e30f;
      for (int b = b0; b < b0 + 128; ++b)
        m = fmaxf(m, sx[b * N_ + n] + s_lds[b]);
      red_lds[half * N_ + n] = m;
      __syncthreads();
      m = fmaxf(red_lds[n], red_lds[N_ + n]);
      float ssum = 0.f;
      for (int b = b0; b < b0 + 128; ++b)
        ssum += __expf(sx[b * N_ + n] + s_lds[b] - m);
      __syncthreads();
      red_lds[half * N_ + n] = ssum;
      __syncthreads();
      const float inv = 1.f / (red_lds[n] + red_lds[N_ + n]);
      for (int b = b0; b < b0 + 128; ++b) {
        float a = __expf(sx[b * N_ + n] + s_lds[b] - m) * inv;
        float w = a * input[((size_t)b * T1_ + t) * N_ + n];
        A_lds[b * AST + n] = (__bf16)w;
        if (wg == 0) out_w[((size_t)b * T1_ + t) * N_ + n] = w;
      }
    }

    /* drain the TDM transfer before anyone touches the h columns */
#if __has_builtin(__builtin_amdgcn_tensor_load_to_lds)
    if (wave == 0) __builtin_amdgcn_s_wait_tensorcnt(0);
#endif
    __syncthreads();

    /* GEMM: D[256 x 64] = A[256 x 384] * W^T, bf16 WMMA, f32 accum */
    v8f acc[2][4];
#pragma unroll
    for (int m = 0; m < 2; ++m)
#pragma unroll
      for (int g = 0; g < 4; ++g)
#pragma unroll
        for (int r = 0; r < 8; ++r) acc[m][g][r] = 0.f;

    const int m0    = wave * 2;
    const int arow0 = (m0 * 16 + l16) * AST;
    const int arow1 = arow0 + 16 * AST;
    const int akoff = lhalf * 8;   /* A lane layout: K {0..7,16..23} / {8..15,24..31} */
    const int bkoff = lhalf * 8;   /* B lane layout: K {0..7,16..23} / {8..15,24..31} */

#pragma unroll
    for (int kk = 0; kk < K_; kk += 32) {
      v16bf a0, a1, bfr[4];
      {
        v8bf lo = *(const v8bf*)(&A_lds[arow0 + kk + akoff]);
        v8bf hi = *(const v8bf*)(&A_lds[arow0 + kk + akoff + 16]);
        a0 = __builtin_shufflevector(lo, hi, 0,1,2,3,4,5,6,7,8,9,10,11,12,13,14,15);
      }
      {
        v8bf lo = *(const v8bf*)(&A_lds[arow1 + kk + akoff]);
        v8bf hi = *(const v8bf*)(&A_lds[arow1 + kk + akoff + 16]);
        a1 = __builtin_shufflevector(lo, hi, 0,1,2,3,4,5,6,7,8,9,10,11,12,13,14,15);
      }
#pragma unroll
      for (int g = 0; g < 4; ++g) {
        v8bf lo = *(const v8bf*)(&W_lds[(g * 16 + l16) * AST + kk + bkoff]);
        v8bf hi = *(const v8bf*)(&W_lds[(g * 16 + l16) * AST + kk + bkoff + 16]);
        bfr[g] = __builtin_shufflevector(lo, hi, 0,1,2,3,4,5,6,7,8,9,10,11,12,13,14,15);
      }
#pragma unroll
      for (int g = 0; g < 4; ++g) {
        acc[0][g] = __builtin_amdgcn_wmma_f32_16x16x32_bf16(false, a0, false, bfr[g],
                                                            (short)0, acc[0][g], false, false);
        acc[1][g] = __builtin_amdgcn_wmma_f32_16x16x32_bf16(false, a1, false, bfr[g],
                                                            (short)0, acc[1][g], false, false);
      }
    }

    /* LSTM pointwise; publish h (bf16) + encoded output + s partial */
    __bf16* hdst = hbf + (size_t)nxt * B_ * H_;
    float*  spw  = spart + (size_t)nxt * NWG * B_ + (size_t)wg * B_;
#pragma unroll
    for (int m = 0; m < 2; ++m) {
#pragma unroll
      for (int r = 0; r < 8; ++r) {
        const int b = (m0 + m) * 16 + r + lhalf * 8;
        float iv = acc[m][0][r] + bsum[0];
        float fv = acc[m][1][r] + bsum[1];
        float gv = acc[m][2][r] + bsum[2];
        float ov = acc[m][3][r] + bsum[3];
        float ig = 1.f / (1.f + __expf(-iv));
        float fg = 1.f / (1.f + __expf(-fv));
        float gg = tanhf(gv);
        float og = 1.f / (1.f + __expf(-ov));
        float cn = fg * cReg[m][r] + ig * gg;
        float hn = og * tanhf(cn);
        cReg[m][r] = cn;
        hdst[(size_t)b * H_ + jglob] = (__bf16)hn;
        out_e[((size_t)b * T1_ + t) * H_ + jglob] = hn;
        /* deterministic 16-lane reduction over owned j columns */
        float v = hn * Whj + cn * Wcj;
        v += __shfl_xor(v, 1);
        v += __shfl_xor(v, 2);
        v += __shfl_xor(v, 4);
        v += __shfl_xor(v, 8);
        if (l16 == 0) s_lds[b] = v;
      }
    }
    __syncthreads();
    spw[tid] = s_lds[tid];

    /* ---- one device-wide barrier per step ---- */
    __threadfence();
    __syncthreads();
    __builtin_amdgcn_s_cluster_barrier();    /* s_barrier_signal/-wait -3 (NOP w/o cluster) */
    if (tid == 0) {
      atomicAdd(cnt, 1u);
      const unsigned target = (unsigned)(NWG * (t + 1));
      while (__hip_atomic_load(cnt, __ATOMIC_ACQUIRE, __HIP_MEMORY_SCOPE_AGENT) < target)
        __builtin_amdgcn_s_sleep(1);
    }
    __syncthreads();
    __threadfence();
  }
}

extern "C" void kernel_launch(void* const* d_in, const int* in_sizes, int n_in,
                              void* d_out, int out_size, void* d_ws, size_t ws_size,
                              hipStream_t stream) {
  const float* input  = (const float*)d_in[0];
  const float* attn_W = (const float*)d_in[1];
  const float* attn_b = (const float*)d_in[2];
  const float* W_ih   = (const float*)d_in[3];
  const float* W_hh   = (const float*)d_in[4];
  const float* b_ih   = (const float*)d_in[5];
  const float* b_hh   = (const float*)d_in[6];
  unsigned char* ws   = (unsigned char*)d_ws;
  float* out          = (float*)d_out;

  darnn_prep<<<1536, 256, 0, stream>>>(input, attn_W, attn_b, W_ih, W_hh, ws);
  darnn_main<<<NWG, NTHR, 0, stream>>>(input, attn_W, b_ih, b_hh, ws, out);
}
